// SparseConv2D_33251636806221
// MI455X (gfx1250) — matrix operand order repaired and checked by reading, not compiled, not run on hardware
//
#include <hip/hip_runtime.h>
#include <hip/hip_bf16.h>

// Problem constants (from reference)
#define B_N    8
#define C_IN   64
#define H_IN   112
#define W_IN   112
#define F_OUT  128
#define OH     110
#define OW     110
#define K_TOT  576          // C_IN * 3 * 3
#define KSTEPS 18           // K_TOT / 32
#define HTILES 7            // ceil(OH / 16)
#define K_PAD  584          // 576 + 8 bf16 pad -> 1168 B row stride (16 distinct bank quads)

typedef __attribute__((ext_vector_type(16))) __bf16 v16bf;
typedef __attribute__((ext_vector_type(8)))  __bf16 v8bf;
typedef __attribute__((ext_vector_type(8)))  float  v8f;

// ---------------------------------------------------------------------------
// Pass 1: pack f32 weights [F, 576] -> bf16 [F, 576] in workspace.
// ---------------------------------------------------------------------------
__global__ __launch_bounds__(256) void pack_weights_bf16(
    const float* __restrict__ k32, __bf16* __restrict__ kb, int n)
{
    int i = blockIdx.x * 256 + threadIdx.x;
    if (i < n) kb[i] = (__bf16)k32[i];
}

// ---------------------------------------------------------------------------
// Pass 2: im2col-GEMM with V_WMMA_F32_16X16X32_BF16.
// Block = 256 threads = 8 waves = the 8 f-tiles of F=128; one block owns a
// 16-column (h) output slab for fixed (b, w).  The im2col panel for those 16
// columns is built cooperatively in LDS as bf16 once, then each wave streams
// 18 WMMAs against it.
// ---------------------------------------------------------------------------
__global__ __launch_bounds__(256) void sconv2d_wmma_bf16(
    const float*  __restrict__ x,     // [B, C, H, W] f32
    const __bf16* __restrict__ wA,    // [F, 576] bf16 (packed)
    float*        __restrict__ out)   // [B, F, OW, OH] f32
{
    __shared__ __bf16 panel[16][K_PAD];          // 18,688 B

    int bt = blockIdx.x;
    const int ht = bt % HTILES;  bt /= HTILES;   // h tile (0..6)
    const int w  = bt % OW;      bt /= OW;       // output w (0..109)
    const int b  = bt;                           // batch
    const int h0 = ht * 16;

    const int nlane = threadIdx.x & 15;          // column within tile -> h
    const int hfill = min(h0 + nlane, OH - 1);   // clamped h for loads

    // ---- cooperative im2col fill: 16 threads per column, k strided by 16.
    //      k = (i*3 + j)*64 + c ;  value = x[b, c, h+j, w+i]
    {
        const float* xb = x + ((size_t)b * C_IN * H_IN + hfill) * W_IN + w;
        for (int k = (threadIdx.x >> 4); k < K_TOT; k += 16) {
            const int p  = k >> 6;               // i*3 + j
            const int c  = k & 63;
            const int ci = p / 3;                // offset along W
            const int cj = p % 3;                // offset along H
            panel[nlane][k] =
                (__bf16)xb[(size_t)c * (H_IN * W_IN) + (size_t)cj * W_IN + ci];
        }
    }
    __syncthreads();

    // ---- per-wave GEMM: 16(F) x 16(h) tile
    const int wave = threadIdx.x >> 5;           // 0..7  -> f tile
    const int lane = threadIdx.x & 31;
    const int f0   = wave * 16;
    const int hi   = lane >> 4;                  // lane half (0/1)
    const int kb   = hi * 16;                    // A-matrix K sub-offset

    const int h      = h0 + nlane;
    const bool valid = (h < OH);

    // A: this lane's weight row (M = f0 + nlane), K chunk base kb
    const __bf16* arow = wA + (size_t)(f0 + nlane) * K_TOT + kb;
    // B: lane half hi holds K = k0 + 16*hi + e  (e = 0..15), contiguous in LDS
    const __bf16* bcol = &panel[nlane][hi * 16];

    v8f acc = {};

    #pragma unroll
    for (int s = 0; s < KSTEPS; ++s) {
        const int k0 = s * 32;

        // A fragment: two contiguous 16B global chunks (contiguous k = 16*hi + e, same order as B)
        v8bf a0 = *(const v8bf*)(arow + k0);         // K = k0+kb    .. +7
        v8bf a1 = *(const v8bf*)(arow + k0 + 8);     // K = k0+kb+8  .. +15
        v16bf afr;
        #pragma unroll
        for (int t = 0; t < 8; ++t) { afr[t] = a0[t]; afr[t + 8] = a1[t]; }

        // B fragment: 32 contiguous bytes from LDS (2x ds_load_b128)
        v16bf bfr = *(const v16bf*)(bcol + k0);

        acc = __builtin_amdgcn_wmma_f32_16x16x32_bf16(
            /*neg_a=*/false, afr, /*neg_b=*/false, bfr,
            /*c_mod=*/(short)0, acc, /*reuse_a=*/false, /*reuse_b=*/false);
    }

    // ---- store: lane column N = h; VGPR r holds M = r + 8*hi
    if (valid) {
        float* obase = out + ((size_t)b * F_OUT * OW + w) * OH + h;
        #pragma unroll
        for (int r = 0; r < 8; ++r) {
            const int f = f0 + r + 8 * hi;
            obase[(size_t)f * (OW * OH)] = acc[r];
        }
    }
}

// ---------------------------------------------------------------------------
extern "C" void kernel_launch(void* const* d_in, const int* in_sizes, int n_in,
                              void* d_out, int out_size, void* d_ws, size_t ws_size,
                              hipStream_t stream)
{
    const float* x   = (const float*)d_in[0];   // inputs [8,64,112,112] f32
    const float* k32 = (const float*)d_in[1];   // kernel [128,576] f32
    float* out = (float*)d_out;                 // [8,128,110,110] f32
    __bf16* wb = (__bf16*)d_ws;                 // 128*576*2 = 147456 bytes

    const int nW = F_OUT * K_TOT;
    pack_weights_bf16<<<(nW + 255) / 256, 256, 0, stream>>>(k32, wb, nW);

    // one block = 8 waves = 8 f-tiles; grid covers (b, w, h-tile)
    dim3 grid(B_N * OW * HTILES);               // 8*110*7 = 6160 blocks
    sconv2d_wmma_bf16<<<grid, 256, 0, stream>>>(x, wb, out);
}
